// Net_conpu_v7_8005819040197
// MI455X (gfx1250) — matrix operand order repaired and checked by reading, not compiled, not run on hardware
//
#include <hip/hip_runtime.h>
#include <hip/hip_bf16.h>

typedef __bf16 bf16_t;
typedef __attribute__((ext_vector_type(16))) __bf16 v16bf;
typedef __attribute__((ext_vector_type(8)))  __bf16 v8bf;
typedef __attribute__((ext_vector_type(8)))  float  v8f;
typedef __attribute__((ext_vector_type(4)))  float  v4f;

// ---------------------------------------------------------------------------
// WMMA helpers (CDNA5: D = A(16x32 bf16) x B(32x16 bf16) + C(16x16 f32))
// ---------------------------------------------------------------------------
__device__ __forceinline__ v16bf cat8(v8bf a, v8bf b) {
  return __builtin_shufflevector(a, b, 0, 1, 2, 3, 4, 5, 6, 7,
                                       8, 9, 10, 11, 12, 13, 14, 15);
}

// A tile 16x32 (MxK) bf16, contiguous k order used here (not AMD's table order):
//   lanes 0-15  hold row M=lane,    elems = K {0..15}
//   lanes 16-31 hold row M=lane-16, elems = K {16..31}
__device__ __forceinline__ v16bf load_a_tile(const bf16_t* base, int rowStride,
                                             int row, int kk, int lane) {
  int half8 = (lane >> 4) << 4;                    // 0 or 16
  const bf16_t* p = base + (long)row * rowStride + kk + half8;
  v8bf lo = *(const v8bf*)(p);                     // K = kk+half8 .. +7
  v8bf hi = *(const v8bf*)(p + 8);                 // K = kk+half8+8 .. +15
  return cat8(lo, hi);
}

// B tile 32x16 (KxN) bf16, same contiguous k order as A (not AMD's table order):
//   lanes 0-15  hold col N=lane,    elems = K {0..15}
//   lanes 16-31 hold col N=lane-16, elems = K {16..31}
__device__ __forceinline__ v16bf load_b_tile(const bf16_t* base, int rowStride,
                                             int row, int kk, int lane) {
  int half16 = (lane >> 4) << 4;                   // 0 or 16
  const bf16_t* p = base + (long)row * rowStride + kk + half16;
  v8bf lo = *(const v8bf*)(p);
  v8bf hi = *(const v8bf*)(p + 8);
  return cat8(lo, hi);
}

__device__ __forceinline__ v8f wmma_bf16(v16bf a, v16bf b, v8f c) {
  return __builtin_amdgcn_wmma_f32_16x16x32_bf16(false, a, false, b,
                                                 (short)0, c, false, false);
}

// Branch-free sorted insert into a descending top-4 list (pure dataflow ->
// v_cmp + v_cndmask, no exec-mask juggling).
__device__ __forceinline__ void top4_insert(float (&bv)[4], int (&bi)[4],
                                            float nv, int ni) {
  bool c0 = nv > bv[0];
  bool c1 = nv > bv[1];
  bool c2 = nv > bv[2];
  bool c3 = nv > bv[3];
  float o0 = c0 ? nv : bv[0];
  int   i0 = c0 ? ni : bi[0];
  float o1 = c0 ? bv[0] : (c1 ? nv : bv[1]);
  int   i1 = c0 ? bi[0] : (c1 ? ni : bi[1]);
  float o2 = c1 ? bv[1] : (c2 ? nv : bv[2]);
  int   i2 = c1 ? bi[1] : (c2 ? ni : bi[2]);
  float o3 = c2 ? bv[2] : (c3 ? nv : bv[3]);
  int   i3 = c2 ? bi[2] : (c3 ? ni : bi[3]);
  bv[0] = o0; bv[1] = o1; bv[2] = o2; bv[3] = o3;
  bi[0] = i0; bi[1] = i1; bi[2] = i2; bi[3] = i3;
}

// ---------------------------------------------------------------------------
// Small prep kernels
// ---------------------------------------------------------------------------
__global__ void f32_to_bf16_kernel(const float* __restrict__ src,
                                   bf16_t* __restrict__ dst, int n) {
  int i = blockIdx.x * blockDim.x + threadIdx.x;
  if (i < n) dst[i] = (bf16_t)src[i];
}

// W1 is (64,6): feat = [nbr(3), ctr(3)] -> pad each part to 32 channels.
__global__ void pad_w1_kernel(const float* __restrict__ W1,
                              bf16_t* __restrict__ W1p) {
  int i = blockIdx.x * blockDim.x + threadIdx.x;   // 64*64
  if (i >= 64 * 64) return;
  int o = i >> 6, k = i & 63;
  float v = 0.0f;
  if (k < 3) v = W1[o * 6 + k];
  else if (k >= 32 && k < 35) v = W1[o * 6 + 3 + (k - 32)];
  W1p[i] = (bf16_t)v;
}

// x (B,3,N) f32 channel-major -> X0 (B,N,32) bf16 point-major, zero-padded.
__global__ void prep_x0_kernel(const float* __restrict__ x,
                               bf16_t* __restrict__ X0, int N, int B) {
  int tid = blockIdx.x * blockDim.x + threadIdx.x;
  if (tid >= B * N) return;
  int b = tid / N, n = tid % N;
  const float* xb = x + (long)b * 3 * N;
  bf16_t* o = X0 + (long)tid * 32;
  o[0] = (bf16_t)xb[n];
  o[1] = (bf16_t)xb[N + n];
  o[2] = (bf16_t)xb[2 * N + n];
  #pragma unroll
  for (int c = 3; c < 32; ++c) o[c] = (bf16_t)0.0f;
}

// sq[b,n] = sum_c X[b,n,c]^2
__global__ void sqnorm_kernel(const bf16_t* __restrict__ X, long batchStride,
                              int rowStride, int C, float* __restrict__ sq,
                              int N, int B) {
  int tid = blockIdx.x * blockDim.x + threadIdx.x;
  if (tid >= B * N) return;
  int b = tid / N, n = tid % N;
  const bf16_t* p = X + (long)b * batchStride + (long)n * rowStride;
  float s = 0.0f;
  for (int c = 0; c < C; ++c) { float v = (float)p[c]; s += v * v; }
  sq[tid] = s;
}

// ---------------------------------------------------------------------------
// KNN: per wave, 16 query rows x all N cols. A chunks cached in registers;
// each column tile loads only B. pd = 2*inner - sq_r - sq_c; per-lane top-4,
// then LDS merge across the 16 lanes sharing a row.
// ---------------------------------------------------------------------------
template <int CPAD>
__global__ __launch_bounds__(32) void knn_kernel(
    const bf16_t* __restrict__ X, long batchStride, int rowStride,
    const float* __restrict__ sq, int* __restrict__ idx, int N) {
  constexpr int NCHUNK = CPAD / 32;
  int nt = N / 16;
  int b = blockIdx.x / nt;
  int rbase = (blockIdx.x % nt) * 16;
  int lane = threadIdx.x;
  int col16 = lane & 15;
  int rowHi = (lane >> 4) << 3;                    // 0 or 8

  const bf16_t* Xb = X + (long)b * batchStride;
  const float* sqb = sq + (long)b * N;

  // Cache the 16 query rows (A tiles) in registers.
  v16bf aReg[NCHUNK];
  #pragma unroll
  for (int c = 0; c < NCHUNK; ++c)
    aReg[c] = load_a_tile(Xb, rowStride, rbase + col16, c * 32, lane);

  float bestv[8][4];
  int   besti[8][4];
  #pragma unroll
  for (int v = 0; v < 8; ++v)
    #pragma unroll
    for (int j = 0; j < 4; ++j) { bestv[v][j] = -3.0e38f; besti[v][j] = 0; }

  float sqRow[8];
  #pragma unroll
  for (int v = 0; v < 8; ++v) sqRow[v] = sqb[rbase + rowHi + v];

  for (int ct = 0; ct < nt; ++ct) {
    int cbase = ct * 16;
    v8f acc = {};
    #pragma unroll
    for (int c = 0; c < NCHUNK; ++c) {
      v16bf bm = load_b_tile(Xb, rowStride, cbase + col16, c * 32, lane);
      acc = wmma_bf16(aReg[c], bm, acc);
    }
    float sqCol = sqb[cbase + col16];
    int cIdx = cbase + col16;
    #pragma unroll
    for (int v = 0; v < 8; ++v) {
      float nv = 2.0f * acc[v] - sqRow[v] - sqCol;
      top4_insert(bestv[v], besti[v], nv, cIdx);
    }
  }

  __shared__ float lv[16][16][4];
  __shared__ int   li[16][16][4];
  #pragma unroll
  for (int v = 0; v < 8; ++v)
    #pragma unroll
    for (int j = 0; j < 4; ++j) {
      lv[rowHi + v][col16][j] = bestv[v][j];
      li[rowHi + v][col16][j] = besti[v][j];
    }
  __syncthreads();

  if (lane < 16) {
    float tv[4]; int ti[4];
    #pragma unroll
    for (int j = 0; j < 4; ++j) { tv[j] = -3.0e38f; ti[j] = 0; }
    for (int s = 0; s < 16; ++s)
      #pragma unroll
      for (int j = 0; j < 4; ++j)
        top4_insert(tv, ti, lv[lane][s][j], li[lane][s][j]);
    int* op = idx + ((long)b * N + rbase + lane) * 4;
    #pragma unroll
    for (int j = 0; j < 4; ++j) op[j] = ti[j];
  }
}

// ---------------------------------------------------------------------------
// Edge conv: one wave owns 16 feat-rows (4 points x 4 neighbors) and loops
// over ALL out-channel tiles with its gathered A chunks cached in registers.
// feat row = [X[nbr], X[ctr]]; chunks never straddle the concat boundary.
// ReLU + max over 4-neighbor row groups; store bf16 into the feats slice.
// ---------------------------------------------------------------------------
template <int CPART>
__global__ __launch_bounds__(32) void edgeconv_kernel(
    const bf16_t* __restrict__ Xin, long batchStride, int rowStride,
    const bf16_t* __restrict__ Wb, int O, const int* __restrict__ idx,
    bf16_t* __restrict__ feats, int choff, int N) {
  constexpr int K = 2 * CPART;
  constexpr int NCHUNK = K / 32;
  int pt = N / 4;
  int b = blockIdx.x / pt;
  int pbase = (blockIdx.x % pt) * 4;
  int lane = threadIdx.x;
  int col16 = lane & 15;

  const bf16_t* Xb = Xin + (long)b * batchStride;
  const int* idxb = idx + ((long)b * N + pbase) * 4;   // 16 consecutive ints

  int p = pbase + (col16 >> 2);                        // this A-row's point
  int nbrRow = idxb[col16];                            // this A-row's neighbor

  // Gather the 16 edge-feature rows once, cached across all out tiles.
  v16bf aReg[NCHUNK];
  #pragma unroll
  for (int c = 0; c < NCHUNK; ++c) {
    int kk = c * 32;
    int part = (kk >= CPART);                          // 0 = nbr, 1 = ctr
    int cb = kk - part * CPART;
    int srcRow = part ? p : nbrRow;
    aReg[c] = load_a_tile(Xb, rowStride, srcRow, cb, lane);
  }

  int p0 = pbase + ((lane >> 4) ? 2 : 0);
  long outRow = ((long)b * N + p0) * 512 + choff + col16;

  for (int ot = 0; ot < O / 16; ++ot) {
    int obase = ot * 16;
    v8f acc = {};
    #pragma unroll
    for (int c = 0; c < NCHUNK; ++c) {
      v16bf w = load_b_tile(Wb, K, obase + col16, c * 32, lane);
      acc = wmma_bf16(aReg[c], w, acc);
    }
    // D rows: lanes<16 -> M 0..7 (points pbase+0,+1); lanes>=16 -> M 8..15.
    float m0 = fmaxf(fmaxf(acc[0], acc[1]), fmaxf(acc[2], acc[3]));
    float m1 = fmaxf(fmaxf(acc[4], acc[5]), fmaxf(acc[6], acc[7]));
    m0 = fmaxf(m0, 0.0f);
    m1 = fmaxf(m1, 0.0f);
    feats[outRow + obase]       = (bf16_t)m0;
    feats[outRow + obase + 512] = (bf16_t)m1;
  }
}

// ---------------------------------------------------------------------------
// Final GEMM: out[b,o,n] = sum_c W5[o,c] * feats[b,n,c].
// One wave = 16 points x 8 out-tiles (128 embeddings); A (16x512) cached
// in registers, reused across the 8 out tiles.
// ---------------------------------------------------------------------------
template <int CCH>
__global__ __launch_bounds__(32) void final_gemm_kernel(
    const bf16_t* __restrict__ feats, const bf16_t* __restrict__ W5b,
    float* __restrict__ out, int N, int E) {
  constexpr int NCHUNK = CCH / 32;                 // 16
  constexpr int OGRP = 8;                          // out tiles per wave
  int nt = N / 16, og = E / (16 * OGRP);
  int wid = blockIdx.x;
  int b = wid / (nt * og);
  int rem = wid % (nt * og);
  int nbase = (rem / og) * 16;
  int ogbase = (rem % og) * 16 * OGRP;
  int lane = threadIdx.x;
  int col16 = lane & 15;

  const bf16_t* Fb = feats + (long)b * N * CCH;

  v16bf aReg[NCHUNK];
  #pragma unroll
  for (int c = 0; c < NCHUNK; ++c)
    aReg[c] = load_a_tile(Fb, CCH, nbase + col16, c * 32, lane);

  for (int ot = 0; ot < OGRP; ++ot) {
    int obase = ogbase + ot * 16;
    v8f acc = {};
    #pragma unroll
    for (int c = 0; c < NCHUNK; ++c) {
      v16bf w = load_b_tile(W5b, CCH, obase + col16, c * 32, lane);
      acc = wmma_bf16(aReg[c], w, acc);
    }
    int o = obase + col16;
    long outBase = ((long)b * E + o) * N + nbase + ((lane >> 4) ? 8 : 0);
    v4f lo = {acc[0], acc[1], acc[2], acc[3]};
    v4f hi = {acc[4], acc[5], acc[6], acc[7]};
    *(v4f*)(out + outBase)     = lo;
    *(v4f*)(out + outBase + 4) = hi;
  }
}

// ---------------------------------------------------------------------------
// Host launcher
// ---------------------------------------------------------------------------
extern "C" void kernel_launch(void* const* d_in, const int* in_sizes, int n_in,
                              void* d_out, int out_size, void* d_ws,
                              size_t ws_size, hipStream_t stream) {
  const float* x  = (const float*)d_in[0];  // (8, 3, 2048)
  const float* W1 = (const float*)d_in[1];  // (64, 6)
  const float* W2 = (const float*)d_in[2];  // (64, 128)
  const float* W3 = (const float*)d_in[3];  // (128, 128)
  const float* W4 = (const float*)d_in[4];  // (256, 256)
  const float* W5 = (const float*)d_in[5];  // (512, 512)
  float* out = (float*)d_out;               // (8, 512, 2048)

  const int B = 8, N = 2048, E = 512;

  char* ws = (char*)d_ws;
  auto carve = [&](size_t bytes) {
    char* p = ws;
    ws += (bytes + 255) & ~(size_t)255;
    return p;
  };
  bf16_t* X0    = (bf16_t*)carve((size_t)B * N * 32 * sizeof(bf16_t));
  bf16_t* feats = (bf16_t*)carve((size_t)B * N * 512 * sizeof(bf16_t));
  float*  sq    = (float*)carve((size_t)B * N * sizeof(float));
  int*    idx   = (int*)carve((size_t)B * N * 4 * sizeof(int));
  bf16_t* W1p   = (bf16_t*)carve(64 * 64 * sizeof(bf16_t));
  bf16_t* W2b   = (bf16_t*)carve(64 * 128 * sizeof(bf16_t));
  bf16_t* W3b   = (bf16_t*)carve(128 * 128 * sizeof(bf16_t));
  bf16_t* W4b   = (bf16_t*)carve(256 * 256 * sizeof(bf16_t));
  bf16_t* W5b   = (bf16_t*)carve(512 * 512 * sizeof(bf16_t));

  // Weight conversion + input layout prep
  pad_w1_kernel<<<(64 * 64 + 255) / 256, 256, 0, stream>>>(W1, W1p);
  f32_to_bf16_kernel<<<(64 * 128 + 255) / 256, 256, 0, stream>>>(W2, W2b, 64 * 128);
  f32_to_bf16_kernel<<<(128 * 128 + 255) / 256, 256, 0, stream>>>(W3, W3b, 128 * 128);
  f32_to_bf16_kernel<<<(256 * 256 + 255) / 256, 256, 0, stream>>>(W4, W4b, 256 * 256);
  f32_to_bf16_kernel<<<(512 * 512 + 255) / 256, 256, 0, stream>>>(W5, W5b, 512 * 512);
  prep_x0_kernel<<<(B * N + 255) / 256, 256, 0, stream>>>(x, X0, N, B);

  const long fbStride = (long)N * 512;   // feats batch stride (elems)
  const int  pn = (B * N + 255) / 256;
  const int  knnGrid = B * (N / 16);
  const int  convGrid = B * (N / 4);

  // ---- Block 1: X0 (Cpad=32) -> feats[:, :, 0:64]
  sqnorm_kernel<<<pn, 256, 0, stream>>>(X0, (long)N * 32, 32, 32, sq, N, B);
  knn_kernel<32><<<knnGrid, 32, 0, stream>>>(X0, (long)N * 32, 32, sq, idx, N);
  edgeconv_kernel<32><<<convGrid, 32, 0, stream>>>(
      X0, (long)N * 32, 32, W1p, 64, idx, feats, 0, N);

  // ---- Block 2: feats[0:64] (C=64) -> feats[:, :, 64:128]
  sqnorm_kernel<<<pn, 256, 0, stream>>>(feats, fbStride, 512, 64, sq, N, B);
  knn_kernel<64><<<knnGrid, 32, 0, stream>>>(feats, fbStride, 512, sq, idx, N);
  edgeconv_kernel<64><<<convGrid, 32, 0, stream>>>(
      feats, fbStride, 512, W2b, 64, idx, feats, 64, N);

  // ---- Block 3: feats[64:128] (C=64) -> feats[:, :, 128:256]
  sqnorm_kernel<<<pn, 256, 0, stream>>>(feats + 64, fbStride, 512, 64, sq, N, B);
  knn_kernel<64><<<knnGrid, 32, 0, stream>>>(feats + 64, fbStride, 512, sq, idx, N);
  edgeconv_kernel<64><<<convGrid, 32, 0, stream>>>(
      feats + 64, fbStride, 512, W3b, 128, idx, feats, 128, N);

  // ---- Block 4: feats[128:256] (C=128) -> feats[:, :, 256:512]
  sqnorm_kernel<<<pn, 256, 0, stream>>>(feats + 128, fbStride, 512, 128, sq, N, B);
  knn_kernel<128><<<knnGrid, 32, 0, stream>>>(feats + 128, fbStride, 512, sq, idx, N);
  edgeconv_kernel<128><<<convGrid, 32, 0, stream>>>(
      feats + 128, fbStride, 512, W4b, 256, idx, feats, 256, N);

  // ---- Final projection: out = W5 @ feats^T per batch
  final_gemm_kernel<512><<<B * (N / 16) * (E / 128), 32, 0, stream>>>(
      feats, W5b, out, N, E);
}
